// EuclideanCodebook_51049981281494
// MI455X (gfx1250) — matrix operand order repaired and checked by reading, not compiled, not run on hardware
//
#include <hip/hip_runtime.h>

// ---------------- problem constants ----------------
#define N_TOK   32768
#define C_CODE  8192
#define D_DIM   512
#define DECAY   0.8f
#define ONE_M_D 0.2f
#define EPSV    1e-05f

// GEMM tiling
#define BM 128
#define BN 128
#define BK 64
#define NCHUNK (D_DIM / BK)
#define LDA 72   // padded bf16 stride (shorts): 128B data + 16B pad per row

typedef __attribute__((ext_vector_type(16))) __bf16 v16bf;
typedef __attribute__((ext_vector_type(8)))  float  v8f;
typedef __attribute__((ext_vector_type(4)))  unsigned int u32x4;
typedef __attribute__((ext_vector_type(8)))  int i32x8;
typedef __attribute__((ext_vector_type(4)))  int i32x4;

union Frag16 { v16bf v; unsigned int d[8]; };

#if __has_builtin(__builtin_amdgcn_tensor_load_to_lds) && __has_builtin(__builtin_amdgcn_s_wait_tensorcnt)
  #define TDM_AVAILABLE 1
  #if __has_include(<hip/amd_detail/amd_gfx1250_TDM.h>)
    #define TDM_SIX_ARGS 1
  #else
    #define TDM_SIX_ARGS 0
  #endif
#else
  #define TDM_AVAILABLE 0
#endif

static __device__ __forceinline__ unsigned short f2bf(float f) {
    unsigned int u = __float_as_uint(f);
    unsigned int r = (u + 0x7FFFu + ((u >> 16) & 1u)) >> 16;   // RNE
    return (unsigned short)r;
}

#if TDM_AVAILABLE
// Issue one TDM 2D-tile load: tile_dim0=BK cols, tile_dim1=BM rows of bf16,
// row stride D_DIM, LDS padding 4 DWORDs every 32 DWORDs (-> LDA=72 shorts).
static __device__ __forceinline__ void tdm_issue_tile(const unsigned short* gtile,
                                                      unsigned lds_byte_addr,
                                                      unsigned tensor_rows) {
    unsigned long long ga = (unsigned long long)(uintptr_t)gtile;
    u32x4 g0;
    g0[0] = 1u;                                           // count=1, user mode
    g0[1] = lds_byte_addr;                                // LDS dest (bytes)
    g0[2] = (unsigned)(ga & 0xFFFFFFFFu);                 // global_addr lo
    g0[3] = (unsigned)((ga >> 32) & 0x1FFFFFFu)           // global_addr hi
          | (2u << 30);                                   // type=2 (image)
    i32x8 g1;
    g1[0] = (int)((1u << 16)        // data_size = 2 bytes
                | (1u << 20)        // pad_enable
                | (4u << 22)        // pad_interval: 32 DWORDs (=128B row)
                | (3u << 25));      // pad_amount: 4 DWORDs (=16B)
    g1[1] = (int)(((unsigned)D_DIM & 0xFFFFu) << 16);     // tensor_dim0 lo16
    g1[2] = (int)((((unsigned)D_DIM >> 16) & 0xFFFFu)     // tensor_dim0 hi16
                | ((tensor_rows & 0xFFFFu) << 16));       // tensor_dim1 lo16
    g1[3] = (int)(((tensor_rows >> 16) & 0xFFFFu)         // tensor_dim1 hi16
                | ((unsigned)BK << 16));                  // tile_dim0 = 64
    g1[4] = (int)(unsigned)BM;                            // tile_dim1=128, tile_dim2=0
    g1[5] = (int)(unsigned)D_DIM;                         // tensor_dim0_stride lo32
    g1[6] = 0;                                            // stride hi16 | dim1_stride lo16
    g1[7] = 0;
    i32x4 z4 = {0, 0, 0, 0};
#if TDM_SIX_ARGS
    i32x8 z8 = {0, 0, 0, 0, 0, 0, 0, 0};
    __builtin_amdgcn_tensor_load_to_lds(g0, g1, z4, z4, z8, 0);
#else
    __builtin_amdgcn_tensor_load_to_lds(g0, g1, z4, z4, 0);
#endif
}
#endif

// ---------------- f32 -> bf16 pre-conversion (x then embed) ----------------
__global__ __launch_bounds__(256) void convert_kernel(const float* __restrict__ x,
                                                      const float* __restrict__ embed,
                                                      unsigned short* __restrict__ xbf,
                                                      unsigned short* __restrict__ ebf) {
    size_t i = ((size_t)blockIdx.x * 256 + threadIdx.x) * 8;
    const size_t NX = (size_t)N_TOK * D_DIM;
    const float* src;
    unsigned short* dst;
    size_t off;
    if (i < NX) { src = x; dst = xbf; off = i; }
    else        { src = embed; dst = ebf; off = i - NX; }
    float4 a = *(const float4*)(src + off);
    float4 b = *(const float4*)(src + off + 4);
    union { unsigned short u[8]; uint4 v; } o;
    o.u[0] = f2bf(a.x); o.u[1] = f2bf(a.y); o.u[2] = f2bf(a.z); o.u[3] = f2bf(a.w);
    o.u[4] = f2bf(b.x); o.u[5] = f2bf(b.y); o.u[6] = f2bf(b.z); o.u[7] = f2bf(b.w);
    *(uint4*)(dst + off) = o.v;
}

// ---------------- row norms: one wave per row ----------------
__global__ __launch_bounds__(256) void norms_kernel(const float* __restrict__ x,
                                                    const float* __restrict__ embed,
                                                    float* __restrict__ x2,
                                                    float* __restrict__ y2) {
    int w    = (blockIdx.x * 256 + threadIdx.x) >> 5;
    int lane = threadIdx.x & 31;
    if (w >= N_TOK + C_CODE) return;
    const float* p = (w < N_TOK) ? (x + (size_t)w * D_DIM)
                                 : (embed + (size_t)(w - N_TOK) * D_DIM);
    float s = 0.f;
    for (int i = lane * 4; i < D_DIM; i += 128) {
        float4 v = *(const float4*)(p + i);
        s += v.x * v.x + v.y * v.y + v.z * v.z + v.w * v.w;
    }
    for (int m = 16; m > 0; m >>= 1) s += __shfl_xor(s, m, 32);
    if (lane == 0) {
        if (w < N_TOK) x2[w] = s; else y2[w - N_TOK] = s;
    }
}

// ---------------- fused cdist + argmax + gather + scatter ----------------
__global__ __launch_bounds__(256) void cdist_argmax_kernel(
    const unsigned short* __restrict__ xbf, const unsigned short* __restrict__ ebf,
    const float* __restrict__ x, const float* __restrict__ embed,
    const float* __restrict__ x2, const float* __restrict__ y2,
    float* __restrict__ dist_out, float* __restrict__ ind_out,
    float* __restrict__ quant_out, float* __restrict__ bins,
    float* __restrict__ embed_sum)
{
    __shared__ unsigned short xs[2][BM * LDA];
    __shared__ unsigned short es[2][BN * LDA];
    __shared__ unsigned long long bestPacked[BM];
    __shared__ int rowIdx[BM];
    __shared__ float sx2[BM];

    const int tid    = threadIdx.x;
    const int lane   = tid & 31;
    const int wave   = tid >> 5;       // 0..7
    const int wm     = wave & 3;       // 4 waves along M
    const int wn     = wave >> 2;      // 2 waves along N
    const int laneHi = lane >> 4;
    const int laneLo = lane & 15;
    const int n0     = blockIdx.x * BM;

    for (int i = tid; i < BM; i += 256) {
        bestPacked[i] = 0ull;
        sx2[i] = x2[n0 + i];
    }
    __syncthreads();

    float        bestV[2][8];
    unsigned int bestI[2][8];
#pragma unroll
    for (int mi = 0; mi < 2; ++mi)
#pragma unroll
        for (int r = 0; r < 8; ++r) { bestV[mi][r] = -3.4e38f; bestI[mi][r] = 0u; }

    for (int ct = 0; ct < C_CODE / BN; ++ct) {
        const int c0 = ct * BN;
        v8f acc[2][4];
#pragma unroll
        for (int mi = 0; mi < 2; ++mi)
#pragma unroll
            for (int ni = 0; ni < 4; ++ni)
#pragma unroll
                for (int r = 0; r < 8; ++r) acc[mi][ni][r] = 0.f;

        // ---- prologue: fetch chunk 0 into buffer 0 ----
#if TDM_AVAILABLE
        if (wave == 0) {
            tdm_issue_tile(xbf + (size_t)n0 * D_DIM + 0,
                           (unsigned)(uintptr_t)&xs[0][0], (unsigned)N_TOK);
            tdm_issue_tile(ebf + (size_t)c0 * D_DIM + 0,
                           (unsigned)(uintptr_t)&es[0][0], (unsigned)C_CODE);
        }
#else
#pragma unroll
        for (int i = 0; i < 4; ++i) {
            int e = tid + i * 256, row = e >> 3, col = (e & 7) << 3;
            *(uint4*)&xs[0][row * LDA + col] =
                *(const uint4*)(xbf + (size_t)(n0 + row) * D_DIM + col);
            *(uint4*)&es[0][row * LDA + col] =
                *(const uint4*)(ebf + (size_t)(c0 + row) * D_DIM + col);
        }
#endif

        for (int ch = 0; ch < NCHUNK; ++ch) {
            const int buf = ch & 1;
            // ---- fetch next chunk into other buffer (freed at end of ch-1) ----
            if (ch + 1 < NCHUNK) {
                const int kc = (ch + 1) * BK;
#if TDM_AVAILABLE
                if (wave == 0) {
                    tdm_issue_tile(xbf + (size_t)n0 * D_DIM + kc,
                                   (unsigned)(uintptr_t)&xs[buf ^ 1][0], (unsigned)N_TOK);
                    tdm_issue_tile(ebf + (size_t)c0 * D_DIM + kc,
                                   (unsigned)(uintptr_t)&es[buf ^ 1][0], (unsigned)C_CODE);
                }
#else
#pragma unroll
                for (int i = 0; i < 4; ++i) {
                    int e = tid + i * 256, row = e >> 3, col = (e & 7) << 3;
                    *(uint4*)&xs[buf ^ 1][row * LDA + col] =
                        *(const uint4*)(xbf + (size_t)(n0 + row) * D_DIM + kc + col);
                    *(uint4*)&es[buf ^ 1][row * LDA + col] =
                        *(const uint4*)(ebf + (size_t)(c0 + row) * D_DIM + kc + col);
                }
#endif
            }
#if TDM_AVAILABLE
            if (wave == 0) {   // current pair complete (TDM ops retire in order)
                if (ch + 1 < NCHUNK) __builtin_amdgcn_s_wait_tensorcnt(2);
                else                 __builtin_amdgcn_s_wait_tensorcnt(0);
            }
#endif
            __syncthreads();   // publish current buffer to all waves

            // ---- 2 x K=32 WMMA steps on current buffer ----
#pragma unroll
            for (int ks = 0; ks < 2; ++ks) {
                Frag16 a[2], b[4];
#pragma unroll
                for (int mi = 0; mi < 2; ++mi) {
                    int m = wm * 32 + mi * 16 + laneLo;
#pragma unroll
                    for (int v = 0; v < 8; ++v) {
                        int k = ks * 32 + ((v >> 2) << 4) + (laneHi ? 8 : 0) + ((v & 3) << 1);
                        a[mi].d[v] = *(const unsigned int*)&xs[buf][m * LDA + k];
                    }
                }
#pragma unroll
                for (int ni = 0; ni < 4; ++ni) {
                    int n = wn * 64 + ni * 16 + laneLo;
#pragma unroll
                    for (int v = 0; v < 8; ++v) {
                        int k = ks * 32 + ((v >> 2) << 4) + (laneHi ? 8 : 0) + ((v & 3) << 1);
                        b[ni].d[v] = *(const unsigned int*)&es[buf][n * LDA + k];
                    }
                }
#pragma unroll
                for (int mi = 0; mi < 2; ++mi)
#pragma unroll
                    for (int ni = 0; ni < 4; ++ni)
                        acc[mi][ni] = __builtin_amdgcn_wmma_f32_16x16x32_bf16(
                            false, a[mi].v, false, b[ni].v,
                            (short)0, acc[mi][ni], false, false);
            }
            __syncthreads();   // free current buffer for reuse at ch+2
        }

        // ---- epilogue: dist = -sqrt(max(x2 + y2 - 2*xy, 0)); stream + argmax ----
#pragma unroll
        for (int ni = 0; ni < 4; ++ni) {
            int c = c0 + wn * 64 + ni * 16 + laneLo;
            float yy = y2[c];
#pragma unroll
            for (int mi = 0; mi < 2; ++mi) {
                int mbase = wm * 32 + mi * 16 + laneHi * 8;
#pragma unroll
                for (int r = 0; r < 8; ++r) {
                    int m = mbase + r;
                    float d2 = sx2[m] + yy - 2.0f * acc[mi][ni][r];
                    float dv = -__builtin_sqrtf(fmaxf(d2, 0.0f));
                    dist_out[(size_t)(n0 + m) * C_CODE + c] = dv;
                    if (dv > bestV[mi][r] ||
                        (dv == bestV[mi][r] && (unsigned)c < bestI[mi][r])) {
                        bestV[mi][r] = dv; bestI[mi][r] = (unsigned)c;
                    }
                }
            }
        }
    }

    // cross-lane argmax (16 lanes share a row), then cross-wave via packed u64 max
#pragma unroll
    for (int mi = 0; mi < 2; ++mi)
#pragma unroll
        for (int r = 0; r < 8; ++r) {
            float v = bestV[mi][r];
            unsigned idx = bestI[mi][r];
            for (int s = 1; s < 16; s <<= 1) {
                float ov = __shfl_xor(v, s, 32);
                unsigned oi = (unsigned)__shfl_xor((int)idx, s, 32);
                if (ov > v || (ov == v && oi < idx)) { v = ov; idx = oi; }
            }
            if (laneLo == 0) {
                int m = wm * 32 + mi * 16 + laneHi * 8 + r;
                unsigned key = __float_as_uint(v);
                key = (key & 0x80000000u) ? ~key : (key | 0x80000000u);
                unsigned long long pk = ((unsigned long long)key << 32) |
                                        (unsigned long long)(0xFFFFFFFFu - idx);
                atomicMax(&bestPacked[m], pk);
            }
        }
    __syncthreads();

    if (tid < BM) {
        unsigned long long pk = bestPacked[tid];
        int idx = (int)(0xFFFFFFFFu - (unsigned int)(pk & 0xFFFFFFFFull));
        rowIdx[tid] = idx;
        ind_out[n0 + tid] = (float)idx;
        atomicAdd(&bins[idx], 1.0f);
    }
    __syncthreads();

    for (int e = tid; e < BM * D_DIM; e += 256) {
        int row = e >> 9;
        int dc  = e & (D_DIM - 1);
        int idx = rowIdx[row];
        float xv = x[(size_t)(n0 + row) * D_DIM + dc];
        quant_out[(size_t)(n0 + row) * D_DIM + dc] = embed[(size_t)idx * D_DIM + dc];
        atomicAdd(&embed_sum[(size_t)idx * D_DIM + dc], xv);
    }
}

// ---------------- cluster-size EMA + total reduce ----------------
__global__ __launch_bounds__(256) void cluster_kernel(const float* __restrict__ cs,
                                                      const float* __restrict__ bins,
                                                      float* __restrict__ csn_out,
                                                      float* __restrict__ total) {
    __shared__ float red[256];
    int tid = threadIdx.x;
    int i = blockIdx.x * 256 + tid;
    float v = cs[i] * DECAY + bins[i] * ONE_M_D;
    csn_out[i] = v;
    red[tid] = v;
    __syncthreads();
    for (int s = 128; s > 0; s >>= 1) {
        if (tid < s) red[tid] += red[tid + s];
        __syncthreads();
    }
    if (tid == 0) atomicAdd(total, red[0]);
}

// ---------------- embed_avg EMA + laplace normalize ----------------
__global__ __launch_bounds__(256) void finalize_kernel(const float* __restrict__ ea,
                                                       const float* __restrict__ esum,
                                                       const float* __restrict__ csn,
                                                       const float* __restrict__ total,
                                                       float* __restrict__ ean_out,
                                                       float* __restrict__ en_out) {
    size_t i = (size_t)blockIdx.x * 256 + threadIdx.x;
    int c = (int)(i >> 9);
    float ean = ea[i] * DECAY + esum[i] * ONE_M_D;
    ean_out[i] = ean;
    float T = *total;
    float denom = (csn[c] + EPSV) / (T + (float)C_CODE * EPSV) * T;
    en_out[i] = ean / denom;
}

// ---------------- launch ----------------
extern "C" void kernel_launch(void* const* d_in, const int* in_sizes, int n_in,
                              void* d_out, int out_size, void* d_ws, size_t ws_size,
                              hipStream_t stream) {
    const float* x     = (const float*)d_in[0];
    const float* embed = (const float*)d_in[1];
    const float* cs    = (const float*)d_in[2];
    const float* ea    = (const float*)d_in[3];

    float* out   = (float*)d_out;
    float* quant = out;                               // (1,32768,512)
    float* ind   = out + (size_t)16777216;            // (1,32768)
    float* dist  = out + (size_t)16809984;            // (1,32768,8192)
    float* enew  = out + (size_t)285245440;           // (1,8192,512)
    float* csn   = out + (size_t)289439744;           // (1,8192)
    float* ean   = out + (size_t)289447936;           // (1,8192,512)

    float* ws   = (float*)d_ws;
    float* x2   = ws;                                 // 32768
    float* y2   = ws + 32768;                         // 8192
    float* bins = ws + 40960;                         // 8192
    float* tot  = ws + 49152;                         // 1
    float* esum = ws + 65536;                         // 8192*512
    unsigned short* xbf = (unsigned short*)(ws + 4259840);   // 32768*512 bf16
    unsigned short* ebf = xbf + (size_t)16777216;            // 8192*512 bf16

    // zero bins, total, embed_sum accumulators (contiguous range)
    (void)hipMemsetAsync(ws + 40960, 0, (size_t)(65536 - 40960 + 4194304) * sizeof(float), stream);

    convert_kernel<<<10240, 256, 0, stream>>>(x, embed, xbf, ebf);
    norms_kernel<<<(unsigned)(((N_TOK + C_CODE) * 32) / 256), 256, 0, stream>>>(x, embed, x2, y2);
    cdist_argmax_kernel<<<N_TOK / BM, 256, 0, stream>>>(xbf, ebf, x, embed, x2, y2,
                                                        dist, ind, quant, bins, esum);
    cluster_kernel<<<C_CODE / 256, 256, 0, stream>>>(cs, bins, csn, tot);
    finalize_kernel<<<(C_CODE * D_DIM) / 256, 256, 0, stream>>>(ea, esum, csn, tot, ean, enew);
}
